// TokenFusionModule_85409719648343
// MI455X (gfx1250) — matrix operand order repaired and checked by reading, not compiled, not run on hardware
//
#include <hip/hip_runtime.h>
#include <math.h>

typedef float  v2f   __attribute__((ext_vector_type(2)));
typedef float  v8f   __attribute__((ext_vector_type(8)));
typedef __bf16 v8bf  __attribute__((ext_vector_type(8)));
typedef __bf16 v16bf __attribute__((ext_vector_type(16)));

#define DEVINL __device__ __forceinline__

constexpr int C_   = 128;    // DIM
constexpr int HH   = 112;    // H == W
constexpr int HW   = 12544;  // 112*112
constexpr int MP   = 64;     // padded window rows (49 -> 64)
constexpr int NWIN = 2048;   // B * 16 * 16

// ---------------- WMMA helpers (CDNA5, wave32) -----------------------------

DEVINL v8f wmma_f32(v2f a, v2f b, v8f c) {  // exact f32, K=4
  return __builtin_amdgcn_wmma_f32_16x16x4_f32(false, a, false, b, (short)0, c,
                                               false, false);
}
DEVINL v8f wmma_bf(v16bf a, v16bf b, v8f c) {  // bf16, K=32, f32 accum
  return __builtin_amdgcn_wmma_f32_16x16x32_bf16(false, a, false, b, (short)0,
                                                 c, false, false);
}

DEVINL v16bf cat16(v8bf a, v8bf b) {
  return __builtin_shufflevector(a, b, 0, 1, 2, 3, 4, 5, 6, 7, 8, 9, 10, 11,
                                 12, 13, 14, 15);
}

// f32 A/B fragment (16x4): lane<16 row r0+lane ks {k,k+1}; hi lanes {k+2,k+3}
DEVINL v2f frag(const float* src, int ld, int r0, int k) {
  int idx = threadIdx.x & 15, hi = (threadIdx.x >> 4) & 1;
  return *(const v2f*)(src + (size_t)(r0 + idx) * ld + k + (hi << 1));
}

// bf16 A fragment (16x32) from row-major src: ks {k0+hi*8..+7} U {+16}
DEVINL v16bf fragA(const __bf16* src, int ld, int r0, int k0) {
  int idx = threadIdx.x & 15, hi = (threadIdx.x >> 4) & 1;
  const __bf16* p = src + (size_t)(r0 + idx) * ld + k0 + hi * 8;
  return cat16(*(const v8bf*)p, *(const v8bf*)(p + 16));
}
// bf16 B fragment (32x16) with B[k][n] == W[n][k]: ks {k0+hi*8..+7} U {+16}
DEVINL v16bf fragB(const __bf16* src, int ld, int n0, int k0) {
  int idx = threadIdx.x & 15, hi = (threadIdx.x >> 4) & 1;
  const __bf16* p = src + (size_t)(n0 + idx) * ld + k0 + hi * 8;
  return cat16(*(const v8bf*)p, *(const v8bf*)(p + 16));
}

DEVINL void split2(float x, __bf16& h, __bf16& l) {
  __bf16 hh = (__bf16)x;
  h = hh;
  l = (__bf16)(x - (float)hh);
}

// ---------------- K0a: generic f32 -> bf16 hi/lo split ---------------------

__global__ __launch_bounds__(256) void k_split(const float* __restrict__ x,
                                               __bf16* __restrict__ h,
                                               __bf16* __restrict__ l, int n) {
  int i = blockIdx.x * 256 + threadIdx.x;
  if (i >= n) return;
  split2(x[i], h[i], l[i]);
}

// ---------------- K0b: reorder+split c1w (128,256,5,5) ---------------------
// -> w1s[(tap*8+chunk)*128 + co][ci32]  (32 contiguous bf16 per (tap,co))

__global__ __launch_bounds__(256) void k_wrev(const float* __restrict__ c1w,
                                              __bf16* __restrict__ w1sH,
                                              __bf16* __restrict__ w1sL) {
  int t = blockIdx.x * 256 + threadIdx.x;
  if (t >= 25 * 256 * C_) return;
  int tap = t / (256 * C_);
  int rem = t % (256 * C_);
  int ciG = rem / C_;
  int co  = rem % C_;
  size_t dst = ((size_t)(tap * 8 + (ciG >> 5)) * C_ + co) * 32 + (ciG & 31);
  float v = c1w[((size_t)co * 256 + ciG) * 25 + tap];
  split2(v, w1sH[dst], w1sL[dst]);
}

// ---------------- K1: per-window QKV projections (split-bf16) --------------

__global__ __launch_bounds__(256) void k_proj(
    const float* __restrict__ local, const float* __restrict__ G_T,
    const __bf16* __restrict__ qwH, const __bf16* __restrict__ qwL,
    const float* __restrict__ qb, const __bf16* __restrict__ kwH,
    const __bf16* __restrict__ kwL, const float* __restrict__ kb,
    const __bf16* __restrict__ lvwH, const __bf16* __restrict__ lvwL,
    const float* __restrict__ lvb, float* __restrict__ Qb,
    float* __restrict__ Kb, float* __restrict__ Vt) {
  __shared__ __align__(16) __bf16 lwH[MP * C_], lwL[MP * C_];
  __shared__ __align__(16) __bf16 gtH[MP * C_], gtL[MP * C_];
  int win = blockIdx.x;
  int b = win >> 8, wh = (win >> 4) & 15, ww = win & 15;
  int tid = threadIdx.x;

  for (int i = tid; i < MP * C_; i += 256) {
    int q = i >> 7, c = i & 127;
    float lv = 0.f, gv = 0.f;
    if (q < 49) {
      int h = wh * 7 + q / 7, w = ww * 7 + q % 7;
      lv = local[((size_t)(b * C_ + c) * HH + h) * HH + w];
      gv = G_T[(size_t)win * (49 * C_) + q * C_ + c];
    }
    split2(lv, lwH[i], lwL[i]);
    split2(gv, gtH[i], gtL[i]);
  }
  __syncthreads();

  int wave = tid >> 5;
  int m0 = (wave >> 1) * 16;  // 4 m-tiles
  int n0b = (wave & 1) * 64;  // 4 n-tiles each half
  int idx = tid & 15, hi = (tid >> 4) & 1;

  const __bf16* srcH[3] = {lwH, gtH, gtH};
  const __bf16* srcL[3] = {lwL, gtL, gtL};
  const __bf16* wgtH[3] = {qwH, kwH, lvwH};
  const __bf16* wgtL[3] = {qwL, kwL, lvwL};
  const float* bias[3] = {qb, kb, lvb};
#pragma unroll
  for (int job = 0; job < 3; ++job) {
    v8f acc[4] = {};
    for (int k0 = 0; k0 < C_; k0 += 32) {
      v16bf ah = fragA(srcH[job], C_, m0, k0);
      v16bf al = fragA(srcL[job], C_, m0, k0);
#pragma unroll
      for (int t = 0; t < 4; ++t) {
        v16bf bh = fragB(wgtH[job], C_, n0b + t * 16, k0);
        v16bf bl = fragB(wgtL[job], C_, n0b + t * 16, k0);
        acc[t] = wmma_bf(ah, bh, acc[t]);
        acc[t] = wmma_bf(ah, bl, acc[t]);
        acc[t] = wmma_bf(al, bh, acc[t]);
      }
    }
#pragma unroll
    for (int t = 0; t < 4; ++t) {
      int n = n0b + t * 16 + idx;
      float bv = bias[job][n];
#pragma unroll
      for (int r = 0; r < 8; ++r) {
        int m = m0 + r + hi * 8;
        float v = acc[t][r] + bv;
        if (job == 0)
          Qb[(size_t)win * MP * C_ + m * C_ + n] = v;
        else if (job == 1)
          Kb[(size_t)win * MP * C_ + m * C_ + n] = v;
        else
          Vt[(size_t)win * C_ * MP + n * MP + m] = v;
      }
    }
  }
}

// ---------------- K2: attention per window (exact f32 WMMA) ----------------

__global__ __launch_bounds__(256) void k_attn(const float* __restrict__ Qb,
                                              const float* __restrict__ Kb,
                                              const float* __restrict__ Vt,
                                              float* __restrict__ xbuf) {
  __shared__ float bufA[MP * C_];  // Q, then S/P (ld = 64)
  __shared__ float bufB[MP * C_];  // K, then Vt (128 x 64)
  int win = blockIdx.x;
  int b = win >> 8, wh = (win >> 4) & 15, ww = win & 15;
  int tid = threadIdx.x;

  for (int i = tid; i < MP * C_; i += 256) {
    bufA[i] = Qb[(size_t)win * MP * C_ + i];
    bufB[i] = Kb[(size_t)win * MP * C_ + i];
  }
  __syncthreads();

  int wave = tid >> 5;
  int m0 = (wave >> 1) * 16;
  int idx = tid & 15, hi = (tid >> 4) & 1;

  // S = Q K^T  (64x64, K-dim 128)
  int sn0 = (wave & 1) * 32;
  v8f sacc[2] = {};
  for (int k = 0; k < C_; k += 4) {
    v2f a = frag(bufA, C_, m0, k);
#pragma unroll
    for (int t = 0; t < 2; ++t) {
      v2f bf = frag(bufB, C_, sn0 + t * 16, k);
      sacc[t] = wmma_f32(a, bf, sacc[t]);
    }
  }
  __syncthreads();  // all waves done reading Q/K

  // spill S into bufA (ld=64); stage Vt into bufB
#pragma unroll
  for (int t = 0; t < 2; ++t) {
    int n = sn0 + t * 16 + idx;
#pragma unroll
    for (int r = 0; r < 8; ++r) bufA[(m0 + r + hi * 8) * MP + n] = sacc[t][r];
  }
  for (int i = tid; i < C_ * MP; i += 256)
    bufB[i] = Vt[(size_t)win * C_ * MP + i];
  __syncthreads();

  // masked softmax (valid keys j < 49; zero the pad so P@V ignores it)
  if (tid < MP) {
    float* row = bufA + tid * MP;
    float mx = row[0];
    for (int j = 1; j < 49; ++j) mx = fmaxf(mx, row[j]);
    float s = 0.f;
    for (int j = 0; j < 49; ++j) {
      float e = expf(row[j] - mx);
      row[j] = e;
      s += e;
    }
    float inv = 1.f / s;
    for (int j = 0; j < 49; ++j) row[j] *= inv;
    for (int j = 49; j < MP; ++j) row[j] = 0.f;
  }
  __syncthreads();

  // out = P @ V  (64x128, K-dim 64); B[kk=j][n=c] = Vt[c][j]
  int on0 = (wave & 1) * 64;
  v8f oacc[4] = {};
  for (int k = 0; k < MP; k += 4) {
    v2f a = frag(bufA, MP, m0, k);
#pragma unroll
    for (int t = 0; t < 4; ++t) {
      v2f bf = frag(bufB, MP, on0 + t * 16, k);
      oacc[t] = wmma_f32(a, bf, oacc[t]);
    }
  }
  // window_reverse into pixel-major x (B,112,112,128)
#pragma unroll
  for (int t = 0; t < 4; ++t) {
    int n = on0 + t * 16 + idx;
#pragma unroll
    for (int r = 0; r < 8; ++r) {
      int q = m0 + r + hi * 8;
      if (q < 49) {
        int h = wh * 7 + q / 7, w = ww * 7 + q % 7;
        xbuf[((size_t)(b * HH + h) * HH + w) * C_ + n] = oacc[t][r];
      }
    }
  }
}

// ---------------- K3: conv5x5 (Cin=256 concat) + channel-LN, split-bf16 ----

__global__ __launch_bounds__(256) void k_conv5_ln(
    const float* __restrict__ xbuf, const float* __restrict__ glob,
    const __bf16* __restrict__ w1sH, const __bf16* __restrict__ w1sL,
    const float* __restrict__ c1b, const float* __restrict__ lnw,
    const float* __restrict__ lnb, float* __restrict__ y1) {
  // patch stored position-major [pos=12x12][ci=32], ld padded to 40
  __shared__ __align__(16) __bf16 patH[144 * 40];
  __shared__ __align__(16) __bf16 patL[144 * 40];
  __shared__ float ybuf[64 * C_];  // 32 KB
  int blk = blockIdx.x;            // 8 * 14 * 14
  int b = blk / 196;
  int rem = blk % 196;
  int oy = (rem / 14) * 8, ox = (rem % 14) * 8;
  int tid = threadIdx.x;
  int wave = tid >> 5;
  int m0 = (wave >> 1) * 16;
  int n0 = (wave & 1) * 64;
  int idx = tid & 15, hi = (tid >> 4) & 1;
  int mIdx = m0 + idx;
  int py0 = mIdx >> 3, px0 = mIdx & 7;
  int hi8 = hi * 8;

  v8f acc[4] = {};
  for (int chunk = 0; chunk < 8; ++chunk) {
    __syncthreads();
    for (int i = tid; i < 32 * 144; i += 256) {
      int pos = i >> 5, cc = i & 31;
      int gy = oy + pos / 12 - 2, gx = ox + pos % 12 - 2;
      float v = 0.f;
      if (gy >= 0 && gy < HH && gx >= 0 && gx < HH) {
        int ch = chunk * 32 + cc;
        if (ch < C_)
          v = xbuf[((size_t)(b * HH + gy) * HH + gx) * C_ + ch];
        else
          v = glob[((size_t)(b * C_ + (ch - C_)) * HH + gy) * HH + gx];
      }
      split2(v, patH[pos * 40 + cc], patL[pos * 40 + cc]);
    }
    __syncthreads();
    for (int tap = 0; tap < 25; ++tap) {
      int pos = (py0 + tap / 5) * 12 + px0 + tap % 5;
      const __bf16* pH = patH + pos * 40 + hi8;
      const __bf16* pL = patL + pos * 40 + hi8;
      v16bf ah = cat16(*(const v8bf*)pH, *(const v8bf*)(pH + 16));
      v16bf al = cat16(*(const v8bf*)pL, *(const v8bf*)(pL + 16));
      const __bf16* wbH = w1sH + (size_t)(tap * 8 + chunk) * C_ * 32;
      const __bf16* wbL = w1sL + (size_t)(tap * 8 + chunk) * C_ * 32;
#pragma unroll
      for (int t = 0; t < 4; ++t) {
        v16bf bh = fragB(wbH, 32, n0 + t * 16, 0);
        v16bf bl = fragB(wbL, 32, n0 + t * 16, 0);
        acc[t] = wmma_bf(ah, bh, acc[t]);
        acc[t] = wmma_bf(ah, bl, acc[t]);
        acc[t] = wmma_bf(al, bh, acc[t]);
      }
    }
  }
  // bias, stash full channel vectors for LN
#pragma unroll
  for (int t = 0; t < 4; ++t) {
    int n = n0 + t * 16 + idx;
    float bv = c1b[n];
#pragma unroll
    for (int r = 0; r < 8; ++r)
      ybuf[(m0 + r + hi * 8) * C_ + n] = acc[t][r] + bv;
  }
  __syncthreads();
  if (tid < 64) {
    const float* row = ybuf + tid * C_;
    float u = 0.f;
    for (int c = 0; c < C_; ++c) u += row[c];
    u *= (1.f / C_);
    float s = 0.f;
    for (int c = 0; c < C_; ++c) {
      float d = row[c] - u;
      s += d * d;
    }
    float inv = rsqrtf(s * (1.f / C_) + 1e-6f);
    int gy = oy + (tid >> 3), gx = ox + (tid & 7);
    float* dst = y1 + ((size_t)b * HW + gy * HH + gx) * C_;
    for (int c = 0; c < C_; ++c)
      dst[c] = (row[c] - u) * inv * lnw[c] + lnb[c];
  }
}

// ---------------- K4: 1x1 conv 128->512 + exact GELU (split-bf16) ----------

__global__ __launch_bounds__(256) void k_mlp1(
    const float* __restrict__ y1, const __bf16* __restrict__ c2wH,
    const __bf16* __restrict__ c2wL, const float* __restrict__ c2b,
    __bf16* __restrict__ y2H, __bf16* __restrict__ y2L) {
  __shared__ __align__(16) __bf16 xtH[32 * C_], xtL[32 * C_];  // 8 KB each
  int p0 = blockIdx.x * 32;  // pixel base over B*HW
  int tid = threadIdx.x;
  for (int i = tid; i < 32 * C_; i += 256)
    split2(y1[(size_t)p0 * C_ + i], xtH[i], xtL[i]);
  __syncthreads();
  int wave = tid >> 5;
  int m0 = (wave & 1) * 16;
  int n0 = (wave >> 1) * 128;  // 8 n-tiles per wave over 512 channels
  v8f acc[8] = {};
  for (int k0 = 0; k0 < C_; k0 += 32) {
    v16bf ah = fragA(xtH, C_, m0, k0);
    v16bf al = fragA(xtL, C_, m0, k0);
#pragma unroll
    for (int t = 0; t < 8; ++t) {
      v16bf bh = fragB(c2wH, C_, n0 + t * 16, k0);
      v16bf bl = fragB(c2wL, C_, n0 + t * 16, k0);
      acc[t] = wmma_bf(ah, bh, acc[t]);
      acc[t] = wmma_bf(ah, bl, acc[t]);
      acc[t] = wmma_bf(al, bh, acc[t]);
    }
  }
  int idx = tid & 15, hi = (tid >> 4) & 1;
#pragma unroll
  for (int t = 0; t < 8; ++t) {
    int n = n0 + t * 16 + idx;
    float bv = c2b[n];
#pragma unroll
    for (int r = 0; r < 8; ++r) {
      float v = acc[t][r] + bv;
      v = 0.5f * v * (1.f + erff(v * 0.70710678118654752f));  // exact GELU
      size_t o = (size_t)(p0 + m0 + r + hi * 8) * 512 + n;
      split2(v, y2H[o], y2L[o]);
    }
  }
}

// ---------------- K5: 1x1 conv 512->128, NCHW output (split-bf16) ----------

__global__ __launch_bounds__(256) void k_mlp2(
    const __bf16* __restrict__ y2H, const __bf16* __restrict__ y2L,
    const __bf16* __restrict__ c3wH, const __bf16* __restrict__ c3wL,
    const float* __restrict__ c3b, float* __restrict__ out) {
  __shared__ __align__(16) __bf16 xtH[32 * 512], xtL[32 * 512];  // 32 KB each
  int p0 = blockIdx.x * 32;
  int tid = threadIdx.x;
  for (int i = tid; i < 32 * 512; i += 256) {
    xtH[i] = y2H[(size_t)p0 * 512 + i];
    xtL[i] = y2L[(size_t)p0 * 512 + i];
  }
  __syncthreads();
  int wave = tid >> 5;
  int m0 = (wave & 1) * 16;
  int n0 = (wave >> 1) * 32;  // 2 n-tiles per wave over 128 channels
  v8f acc[2] = {};
  for (int k0 = 0; k0 < 512; k0 += 32) {
    v16bf ah = fragA(xtH, 512, m0, k0);
    v16bf al = fragA(xtL, 512, m0, k0);
#pragma unroll
    for (int t = 0; t < 2; ++t) {
      v16bf bh = fragB(c3wH, 512, n0 + t * 16, k0);
      v16bf bl = fragB(c3wL, 512, n0 + t * 16, k0);
      acc[t] = wmma_bf(ah, bh, acc[t]);
      acc[t] = wmma_bf(ah, bl, acc[t]);
      acc[t] = wmma_bf(al, bh, acc[t]);
    }
  }
  int idx = tid & 15, hi = (tid >> 4) & 1;
#pragma unroll
  for (int t = 0; t < 2; ++t) {
    int n = n0 + t * 16 + idx;
    float bv = c3b[n];
#pragma unroll
    for (int r = 0; r < 8; ++r) {
      int p = p0 + m0 + r + hi * 8;
      int b = p / HW, pr = p % HW;
      out[((size_t)(b * C_ + n)) * HW + pr] = acc[t][r] + bv;
    }
  }
}

// ---------------- host launcher --------------------------------------------

extern "C" void kernel_launch(void* const* d_in, const int* in_sizes, int n_in,
                              void* d_out, int out_size, void* d_ws,
                              size_t ws_size, hipStream_t stream) {
  (void)in_sizes; (void)n_in; (void)out_size; (void)ws_size;
  const float* local = (const float*)d_in[0];
  const float* glob  = (const float*)d_in[1];
  const float* G_T   = (const float*)d_in[2];
  const float* qw  = (const float*)d_in[3];
  const float* qb  = (const float*)d_in[4];
  const float* kw  = (const float*)d_in[5];
  const float* kb  = (const float*)d_in[6];
  const float* lvw = (const float*)d_in[7];
  const float* lvb = (const float*)d_in[8];
  const float* c1w = (const float*)d_in[9];
  const float* c1b = (const float*)d_in[10];
  const float* lnw = (const float*)d_in[11];
  const float* lnb = (const float*)d_in[12];
  const float* c2w = (const float*)d_in[13];
  const float* c2b = (const float*)d_in[14];
  const float* c3w = (const float*)d_in[15];
  const float* c3b = (const float*)d_in[16];
  float* out = (float*)d_out;
  char* ws = (char*)d_ws;

  // ws layout (bytes). y2H/y2L alias the Q/K/Vt/x region (dead after K3).
  const size_t OFF_Q   = 0;                    // 64 MiB
  const size_t OFF_K   = 67108864;             // 64 MiB
  const size_t OFF_V   = 134217728;            // 64 MiB
  const size_t OFF_X   = 201326592;            // 51,380,224 B -> end 252,706,816
  const size_t OFF_W1H = 252706816;            // 1,638,400 B
  const size_t OFF_W1L = OFF_W1H + 1638400;
  const size_t OFF_QWH = OFF_W1L + 1638400;    // 6 x 32,768 B
  const size_t OFF_QWL = OFF_QWH + 32768;
  const size_t OFF_KWH = OFF_QWL + 32768;
  const size_t OFF_KWL = OFF_KWH + 32768;
  const size_t OFF_VWH = OFF_KWL + 32768;
  const size_t OFF_VWL = OFF_VWH + 32768;
  const size_t OFF_2WH = OFF_VWL + 32768;      // 2 x 131,072 B
  const size_t OFF_2WL = OFF_2WH + 131072;
  const size_t OFF_3WH = OFF_2WL + 131072;     // 2 x 131,072 B
  const size_t OFF_3WL = OFF_3WH + 131072;
  const size_t OFF_Y1  = OFF_3WL + 131072;     // 51,380,224 B
  const size_t OFF_Y2H = 0;                    // 102,760,448 B (alias)
  const size_t OFF_Y2L = 102760448;            // 102,760,448 B (alias)

  float*  Qb   = (float*)(ws + OFF_Q);
  float*  Kb   = (float*)(ws + OFF_K);
  float*  Vt   = (float*)(ws + OFF_V);
  float*  xb   = (float*)(ws + OFF_X);
  __bf16* w1sH = (__bf16*)(ws + OFF_W1H);
  __bf16* w1sL = (__bf16*)(ws + OFF_W1L);
  __bf16* qwH  = (__bf16*)(ws + OFF_QWH);
  __bf16* qwL  = (__bf16*)(ws + OFF_QWL);
  __bf16* kwH  = (__bf16*)(ws + OFF_KWH);
  __bf16* kwL  = (__bf16*)(ws + OFF_KWL);
  __bf16* lvwH = (__bf16*)(ws + OFF_VWH);
  __bf16* lvwL = (__bf16*)(ws + OFF_VWL);
  __bf16* c2wH = (__bf16*)(ws + OFF_2WH);
  __bf16* c2wL = (__bf16*)(ws + OFF_2WL);
  __bf16* c3wH = (__bf16*)(ws + OFF_3WH);
  __bf16* c3wL = (__bf16*)(ws + OFF_3WL);
  float*  y1   = (float*)(ws + OFF_Y1);
  __bf16* y2H  = (__bf16*)(ws + OFF_Y2H);
  __bf16* y2L  = (__bf16*)(ws + OFF_Y2L);

  // weight prep
  k_split<<<64, 256, 0, stream>>>(qw, qwH, qwL, 16384);
  k_split<<<64, 256, 0, stream>>>(kw, kwH, kwL, 16384);
  k_split<<<64, 256, 0, stream>>>(lvw, lvwH, lvwL, 16384);
  k_split<<<256, 256, 0, stream>>>(c2w, c2wH, c2wL, 65536);
  k_split<<<256, 256, 0, stream>>>(c3w, c3wH, c3wL, 65536);
  k_wrev<<<3200, 256, 0, stream>>>(c1w, w1sH, w1sL);

  // pipeline
  k_proj<<<NWIN, 256, 0, stream>>>(local, G_T, qwH, qwL, qb, kwH, kwL, kb,
                                   lvwH, lvwL, lvb, Qb, Kb, Vt);
  k_attn<<<NWIN, 256, 0, stream>>>(Qb, Kb, Vt, xb);
  k_conv5_ln<<<8 * 14 * 14, 256, 0, stream>>>(xb, glob, w1sH, w1sL, c1b, lnw,
                                              lnb, y1);
  k_mlp1<<<(8 * HW) / 32, 256, 0, stream>>>(y1, c2wH, c2wL, c2b, y2H, y2L);
  k_mlp2<<<(8 * HW) / 32, 256, 0, stream>>>(y2H, y2L, c3wH, c3wL, c3b, out);
}
